// CorrNet_59012850647739
// MI455X (gfx1250) — matrix operand order repaired and checked by reading, not compiled, not run on hardware
//
#include <hip/hip_runtime.h>
#include <hip/hip_bf16.h>
#include <math.h>

typedef __attribute__((ext_vector_type(16))) _Float16 v16h;
typedef __attribute__((ext_vector_type(8)))  float    v8f;
typedef __attribute__((ext_vector_type(4)))  float    v4f;

#define LROW 129                    // padded LDS row stride (129 % 64 == 1 -> conflict-free)
#define NEG_INF (-3.402823466e38f)

__device__ __forceinline__ float leaky(float v){ return v > 0.f ? v : 0.2f*v; }

// out[c] = leaky( b[c] + sum_k in[k] * W[k*Cout+c] )   (row pointers live in LDS)
__device__ __forceinline__ void linear_leaky_row(const float* in, float* out, int Cin, int Cout,
                                                 const float* __restrict__ W,
                                                 const float* __restrict__ b){
  for(int c=0;c<Cout;++c){
    float v=b[c];
    for(int k=0;k<Cin;++k) v = fmaf(in[k], W[(size_t)k*Cout+c], v);
    out[c]=leaky(v);
  }
}

// torch GroupNorm per sample: groups of contiguous channels, biased variance, eps=1e-5
__device__ __forceinline__ void groupnorm_row(float* row, int C, int groups,
                                              const float* __restrict__ gw,
                                              const float* __restrict__ gb){
  int gs=C/groups;
  for(int g=0; g<groups; ++g){
    float s=0.f, ss=0.f;
    for(int k=0;k<gs;++k){ float v=row[g*gs+k]; s+=v; ss+=v*v; }
    float mu = s/(float)gs;
    float var = ss/(float)gs - mu*mu;
    float r = rsqrtf(var + 1e-5f);
    for(int k=0;k<gs;++k){ int c=g*gs+k; row[c]=(row[c]-mu)*r*gw[c]+gb[c]; }
  }
}

__device__ __forceinline__ void atomicMaxF(float* addr, float v){
  unsigned int* ua=(unsigned int*)addr;
  unsigned int old=*ua;
  while(__uint_as_float(old) < v){
    unsigned int assumed=old;
    old=atomicCAS(ua, assumed, __float_as_uint(v));
    if(old==assumed) break;
  }
}

__global__ void k_init(float* gmaxv, float* gmaxp){
  int t=threadIdx.x;
  if(t<128){ gmaxv[t]=NEG_INF; gmaxp[t]=NEG_INF; }
}

// ---------------- PointNet stage 1: 3->32->64->128 (f), hi 128->128 (h), global max(h) ----------
struct P1W {
  const float *W0,*b0,*gw0,*gb0;     // 3->32,  groups 1
  const float *W1,*b1,*gw1,*gb1;     // 32->64, groups 2
  const float *W2,*b2,*gw2,*gb2;     // 64->128,groups 4
  const float *Wh,*bh,*gwh,*gbh;     // 128->128, groups 4
};

__global__ void __launch_bounds__(32)
k_pn1(const float* __restrict__ xyz, int N, P1W p,
      float* __restrict__ fout, float* __restrict__ gmax){
  __shared__ float sA[32*LROW];
  __shared__ float sB[32*LROW];
  int t  = threadIdx.x;           // 0..31, one point per thread (1 wave per block)
  int pt = blockIdx.x*32 + t;
  float* rA=&sA[t*LROW];
  float* rB=&sB[t*LROW];

  float x0=0.f,x1=0.f,x2=0.f;
  if(pt<N){ x0=xyz[3*(size_t)pt]; x1=xyz[3*(size_t)pt+1]; x2=xyz[3*(size_t)pt+2]; }

  for(int c=0;c<32;++c){
    float v = p.b0[c] + x0*p.W0[c] + x1*p.W0[32+c] + x2*p.W0[64+c];
    rA[c]=leaky(v);
  }
  groupnorm_row(rA,32,1,p.gw0,p.gb0);

  linear_leaky_row(rA,rB,32,64,p.W1,p.b1);
  groupnorm_row(rB,64,2,p.gw1,p.gb1);

  linear_leaky_row(rB,rA,64,128,p.W2,p.b2);
  groupnorm_row(rA,128,4,p.gw2,p.gb2);
  if(pt<N){ for(int c=0;c<128;++c) fout[(size_t)pt*128+c]=rA[c]; }

  linear_leaky_row(rA,rB,128,128,p.Wh,p.bh);
  groupnorm_row(rB,128,4,p.gwh,p.gbh);
  if(pt>=N){ for(int c=0;c<128;++c) rB[c]=NEG_INF; }
  __syncthreads();

  for(int c=t;c<128;c+=32){
    float m=NEG_INF;
    for(int r=0;r<32;++r) m=fmaxf(m, sB[r*LROW+c]);
    atomicMaxF(&gmax[c], m);
  }
}

// ---------------- PointNet stage 2: [f|g] -> 128 -> 64 -> y(64) -> L2 normalize ----------------
struct P2W {
  const float *Wy0,*by0,*gwy0,*gby0; // 256->128, groups 4
  const float *Wy1,*by1,*gwy1,*gby1; // 128->64,  groups 2
  const float *yW,*yb;               // 64->64
};

__global__ void __launch_bounds__(32)
k_pn2(const float* __restrict__ f, int N, const float* __restrict__ gmax, P2W p,
      float* __restrict__ outn, _Float16* __restrict__ out16){
  __shared__ float sF[32*LROW];
  __shared__ float sY[32*LROW];
  __shared__ float gterm[128];
  int t  = threadIdx.x;
  int pt = blockIdx.x*32 + t;
  float* rF=&sF[t*LROW];
  float* rY=&sY[t*LROW];

  // g-dependent half of yi0 is the same for every point: gterm[c] = b[c] + sum_k g[k]*W[128+k][c]
  for(int c=t;c<128;c+=32){
    float v=p.by0[c];
    for(int k=0;k<128;++k) v=fmaf(gmax[k], p.Wy0[(size_t)(128+k)*128+c], v);
    gterm[c]=v;
  }

  if(pt<N){ for(int c=0;c<128;++c) rF[c]=f[(size_t)pt*128+c]; }
  else    { for(int c=0;c<128;++c) rF[c]=0.f; }
  __syncthreads();

  for(int c=0;c<128;++c){
    float v=gterm[c];
    for(int k=0;k<128;++k) v=fmaf(rF[k], p.Wy0[(size_t)k*128+c], v);
    rY[c]=leaky(v);
  }
  groupnorm_row(rY,128,4,p.gwy0,p.gby0);

  linear_leaky_row(rY,rF,128,64,p.Wy1,p.by1);
  groupnorm_row(rF,64,2,p.gwy1,p.gby1);

  for(int c=0;c<64;++c){
    float v=p.yb[c];
    for(int k=0;k<64;++k) v=fmaf(rF[k], p.yW[(size_t)k*64+c], v);
    rY[c]=v;                      // final linear, no activation
  }

  float nn=0.f;
  for(int c=0;c<64;++c) nn += rY[c]*rY[c];
  float inv = 1.f / fmaxf(sqrtf(nn), 1e-12f);
  if(pt<N){
    for(int c=0;c<64;++c){
      float d=rY[c]*inv;
      outn [(size_t)pt*64+c]=d;
      out16[(size_t)pt*64+c]=(_Float16)d;
    }
  }
}

// ---------------- similarity GEMM + per-row max/argmax via WMMA f16 ----------------
// Each wave: 2 A row-tiles (32 vf rows) x half of the pf column tiles.
// Ping-pong double-buffered B loads -> branch-free inner loop, 4 WMMA per B fetch.
union HV { v16h v; _Float16 h[16]; v4f q[2]; };

__device__ __forceinline__ void loadB(HV& b0, HV& b1, const _Float16* __restrict__ pf16,
                                      int tile, int l16, int hi){
  // 16-bit B layout (32x16): lanes 0-15 = N with K0..7 / K16..23; lanes 16-31 = N with K8..15 / K24..31 (matches A).
  const _Float16* brow = pf16 + ((size_t)(tile*16 + l16))*64;
  b0.q[0] = *(const v4f*)(brow +      hi*8);
  b0.q[1] = *(const v4f*)(brow + 16 + hi*8);
  b1.q[0] = *(const v4f*)(brow + 32 + hi*8);
  b1.q[1] = *(const v4f*)(brow + 48 + hi*8);
}

#define SIM_STEP(Bb0, Bb1, COLT)                                                                          \
  {                                                                                                       \
    v8f c0 = {}; v8f c1 = {};                                                                             \
    c0 = __builtin_amdgcn_wmma_f32_16x16x32_f16(false, a00.v, false, (Bb0).v, (short)0, c0, false, false);\
    c0 = __builtin_amdgcn_wmma_f32_16x16x32_f16(false, a01.v, false, (Bb1).v, (short)0, c0, false, false);\
    c1 = __builtin_amdgcn_wmma_f32_16x16x32_f16(false, a10.v, false, (Bb0).v, (short)0, c1, false, false);\
    c1 = __builtin_amdgcn_wmma_f32_16x16x32_f16(false, a11.v, false, (Bb1).v, (short)0, c1, false, false);\
    int n_ = (COLT)*16 + l16;                                                                             \
    _Pragma("unroll")                                                                                     \
    for(int i=0;i<8;++i){                                                                                 \
      if(c0[i] > best0[i]){ best0[i]=c0[i]; bidx0[i]=n_; }                                                \
      if(c1[i] > best1[i]){ best1[i]=c1[i]; bidx1[i]=n_; }                                                \
    }                                                                                                     \
  }

__global__ void __launch_bounds__(128)
k_sim(const _Float16* __restrict__ vf16, const _Float16* __restrict__ pf16,
      int colTiles, int nRows, float* __restrict__ pmax, int* __restrict__ pidx){
  int w    = (int)((blockIdx.x*blockDim.x + threadIdx.x) >> 5);  // global wave id
  int pair = w >> 1;            // row-tile pair
  int half = w & 1;             // which half of the column tiles
  int lane = (int)(threadIdx.x & 31);
  int hi   = lane >> 4;
  int l16  = lane & 15;
  int rt0  = pair*2, rt1 = pair*2 + 1;
  int halfTiles = colTiles >> 1;          // 1024 (even)
  int base = half * halfTiles;

  // A tiles held in registers for the whole sweep.
  // 16-bit A layout: lane m (0-15): V0-3 = K0..7, V4-7 = K16..23 ; lane m+16: K8..15 / K24..31.
  HV a00,a01,a10,a11;
  {
    const _Float16* ar0 = vf16 + ((size_t)(rt0*16 + l16))*64;
    const _Float16* ar1 = vf16 + ((size_t)(rt1*16 + l16))*64;
    a00.q[0]=*(const v4f*)(ar0 +      hi*8); a00.q[1]=*(const v4f*)(ar0 + 16 + hi*8);
    a01.q[0]=*(const v4f*)(ar0 + 32 + hi*8); a01.q[1]=*(const v4f*)(ar0 + 48 + hi*8);
    a10.q[0]=*(const v4f*)(ar1 +      hi*8); a10.q[1]=*(const v4f*)(ar1 + 16 + hi*8);
    a11.q[0]=*(const v4f*)(ar1 + 32 + hi*8); a11.q[1]=*(const v4f*)(ar1 + 48 + hi*8);
  }

  float best0[8], best1[8]; int bidx0[8], bidx1[8];
#pragma unroll
  for(int i=0;i<8;++i){ best0[i]=NEG_INF; best1[i]=NEG_INF; bidx0[i]=0; bidx1[i]=0; }

  // Software-pipelined, branch-free: pf16 is padded by one tile so the lookahead
  // load at the last iteration is always in-bounds (loaded but never consumed).
  HV bA0,bA1,bB0,bB1;
  loadB(bA0,bA1,pf16,base,l16,hi);
  for(int i=0;i<halfTiles;i+=2){
    loadB(bB0,bB1,pf16,base+i+1,l16,hi);
    SIM_STEP(bA0,bA1, base+i);
    loadB(bA0,bA1,pf16,base+i+2,l16,hi);
    SIM_STEP(bB0,bB1, base+i+1);
  }

  // Reduce (max, first-index) across the 16 lanes sharing each row. xor masks <16 stay in-half.
#pragma unroll
  for(int m=1;m<16;m<<=1){
#pragma unroll
    for(int i=0;i<8;++i){
      float ov; int oi;
      ov=__shfl_xor(best0[i],m,32); oi=__shfl_xor(bidx0[i],m,32);
      if(ov>best0[i] || (ov==best0[i] && oi<bidx0[i])){ best0[i]=ov; bidx0[i]=oi; }
      ov=__shfl_xor(best1[i],m,32); oi=__shfl_xor(bidx1[i],m,32);
      if(ov>best1[i] || (ov==best1[i] && oi<bidx1[i])){ best1[i]=ov; bidx1[i]=oi; }
    }
  }
  if(l16==0){
#pragma unroll
    for(int i=0;i<8;++i){
      int r0 = rt0*16 + hi*8 + i;   // C layout: vgpr i -> M = i (+8 for upper half-wave)
      int r1 = rt1*16 + hi*8 + i;
      pmax[(size_t)half*nRows + r0]=best0[i];  pidx[(size_t)half*nRows + r0]=bidx0[i];
      pmax[(size_t)half*nRows + r1]=best1[i];  pidx[(size_t)half*nRows + r1]=bidx1[i];
    }
  }
}

// combine the two column-half partials (half-0 indices are always smaller -> strict > keeps first-idx)
__global__ void k_comb(const float* __restrict__ pmax, const int* __restrict__ pidx, int N,
                       float* __restrict__ rowmax, int* __restrict__ rowidx){
  int i=blockIdx.x*blockDim.x+threadIdx.x;
  if(i>=N) return;
  float m0=pmax[i], m1=pmax[(size_t)N+i];
  int   i0=pidx[i], i1=pidx[(size_t)N+i];
  bool take1 = (m1 > m0);
  rowmax[i] = take1 ? m1 : m0;
  rowidx[i] = take1 ? i1 : i0;
}

// ---------------- final tiny MLP: [vf | pf[argmax] | smax](129) -> 64 -> 1 ----------------
__global__ void k_fin(const float* __restrict__ vf, const float* __restrict__ pf,
                      const float* __restrict__ rowmax, const int* __restrict__ rowidx,
                      const float* __restrict__ W1, const float* __restrict__ b1,
                      const float* __restrict__ W2, const float* __restrict__ b2,
                      float* __restrict__ out, int Nv){
  int i = blockIdx.x*blockDim.x + threadIdx.x;
  if(i>=Nv) return;
  const float* v = vf + (size_t)i*64;
  const float* y = pf + (size_t)rowidx[i]*64;
  float smax = rowmax[i];
  float o = b2[0];
  for(int c=0;c<64;++c){
    float s=b1[c];
    for(int k=0;k<64;++k) s=fmaf(v[k], W1[(size_t)k*64+c], s);
    for(int k=0;k<64;++k) s=fmaf(y[k], W1[(size_t)(64+k)*64+c], s);
    s=fmaf(smax, W1[(size_t)128*64+c], s);
    s=fmaxf(s,0.f);                 // ReLU
    o=fmaf(s, W2[c], o);            // fuse second layer; no dynamic reg-array indexing
  }
  out[i]=o;
}

extern "C" void kernel_launch(void* const* d_in, const int* in_sizes, int n_in,
                              void* d_out, int out_size, void* d_ws, size_t ws_size,
                              hipStream_t stream){
  const float* vtx=(const float*)d_in[0];
  const float* pts=(const float*)d_in[1];
  int Nv=in_sizes[0]/3, Np=in_sizes[1]/3;

  // params flattened in setup_inputs() dict insertion order
  P1W p1;
  p1.W0=(const float*)d_in[2];  p1.b0=(const float*)d_in[3];  p1.gw0=(const float*)d_in[4];  p1.gb0=(const float*)d_in[5];
  p1.W1=(const float*)d_in[6];  p1.b1=(const float*)d_in[7];  p1.gw1=(const float*)d_in[8];  p1.gb1=(const float*)d_in[9];
  p1.W2=(const float*)d_in[10]; p1.b2=(const float*)d_in[11]; p1.gw2=(const float*)d_in[12]; p1.gb2=(const float*)d_in[13];
  p1.Wh=(const float*)d_in[14]; p1.bh=(const float*)d_in[15]; p1.gwh=(const float*)d_in[16]; p1.gbh=(const float*)d_in[17];
  P2W p2;
  p2.Wy0=(const float*)d_in[18]; p2.by0=(const float*)d_in[19]; p2.gwy0=(const float*)d_in[20]; p2.gby0=(const float*)d_in[21];
  p2.Wy1=(const float*)d_in[22]; p2.by1=(const float*)d_in[23]; p2.gwy1=(const float*)d_in[24]; p2.gby1=(const float*)d_in[25];
  p2.yW =(const float*)d_in[26]; p2.yb =(const float*)d_in[27];
  const float* fW1=(const float*)d_in[28];
  const float* fb1=(const float*)d_in[29];
  const float* fW2=(const float*)d_in[30];
  const float* fb2=(const float*)d_in[31];

  // workspace layout (~26 MB)
  float* wsf   =(float*)d_ws;
  float* f_vtx = wsf;
  float* f_pts = f_vtx + (size_t)Nv*128;
  float* gmax_v= f_pts + (size_t)Np*128;
  float* gmax_p= gmax_v + 128;
  float* rowmax= gmax_p + 128;
  int*   rowidx=(int*)(rowmax + Nv);
  float* pmax  =(float*)(rowidx + Nv);          // [2][Nv] column-half partial max
  int*   pidx  =(int*)(pmax + (size_t)2*Nv);    // [2][Nv] column-half partial argmax
  _Float16* vf16=(_Float16*)(pidx + (size_t)2*Nv);
  _Float16* pf16=vf16 + (size_t)Nv*64;          // padded by one 16x64 tile (lookahead load)

  float* out_vf=(float*)d_out;
  float* out_pf=out_vf + (size_t)Nv*64;
  float* out_cm=out_pf + (size_t)Np*64;

  k_init<<<1,128,0,stream>>>(gmax_v, gmax_p);
  k_pn1<<<(Nv+31)/32,32,0,stream>>>(vtx,Nv,p1,f_vtx,gmax_v);
  k_pn1<<<(Np+31)/32,32,0,stream>>>(pts,Np,p1,f_pts,gmax_p);
  k_pn2<<<(Nv+31)/32,32,0,stream>>>(f_vtx,Nv,gmax_v,p2,out_vf,vf16);
  k_pn2<<<(Np+31)/32,32,0,stream>>>(f_pts,Np,gmax_p,p2,out_pf,pf16);

  int rowTiles=Nv/16, colTiles=Np/16;            // 512, 2048
  int waves = (rowTiles/2)*2;                    // 2 row-tiles/wave x 2 column halves = 512 waves
  k_sim<<<(waves*32+127)/128,128,0,stream>>>(vf16,pf16,colTiles,Nv,pmax,pidx);
  k_comb<<<(Nv+255)/256,256,0,stream>>>(pmax,pidx,Nv,rowmax,rowidx);
  k_fin<<<(Nv+255)/256,256,0,stream>>>(out_vf,out_pf,rowmax,rowidx,fW1,fb1,fW2,fb2,out_cm,Nv);
}
